// HeteGNN_8710193676511
// MI455X (gfx1250) — matrix operand order repaired and checked by reading, not compiled, not run on hardware
//
#include <hip/hip_runtime.h>
#include <hip/hip_bf16.h>

// ---------------------------------------------------------------------------
// HeteGNN forward for MI455X (gfx1250, wave32, WMMA).
// Dominant cost: 4x edge-gate GEMM [E,96]x[96,96] -> v_wmma_f32_16x16x32_f16
// (f16 in, f32 accumulate). x (9.8MB) is L2-resident; segment-sum via L2
// float atomics. A-fragments register-resident (hoisted out of the nt loop)
// to minimize the ds_load : v_wmma ratio -- LDS is shared by 4 waves/WGP.
// ---------------------------------------------------------------------------

typedef _Float16 h16;
typedef h16   v8h  __attribute__((ext_vector_type(8)));
typedef h16   v16h __attribute__((ext_vector_type(16)));
typedef float v8f  __attribute__((ext_vector_type(8)));

#define N_NODES   25600
#define NSTREAM   12800
#define FDIM      96
#define NE        409600
#define OUTF      128
#define NFLT      (N_NODES * FDIM)   // 2,457,600 floats per node-feature buffer

__device__ __forceinline__ float prelu_f(float x, float a) { return x >= 0.f ? x : a * x; }

__device__ __forceinline__ v16h mk16(v8h lo, v8h hi) {
  return __builtin_shufflevector(lo, hi, 0,1,2,3,4,5,6,7,8,9,10,11,12,13,14,15);
}

// ---------------------------------------------------------------------------
// Kernel 1: grouped 1x1 convs (posture/motion streams) -> x0[N,96] (raw).
// Thread t in [0,25600): s = t/12800 (wave-uniform -> scalar weight loads),
// node n = 2*(t%12800)+s. Fused conv1->PReLU->conv2 without a mid[] array.
// ---------------------------------------------------------------------------
__global__ __launch_bounds__(256)
void k_transform(const float* __restrict__ h,
                 const float* __restrict__ pw1, const float* __restrict__ pb1,
                 const float* __restrict__ pa,  const float* __restrict__ pw2,
                 const float* __restrict__ pb2,
                 const float* __restrict__ mw1, const float* __restrict__ mb1,
                 const float* __restrict__ ma,  const float* __restrict__ mw2,
                 const float* __restrict__ mb2,
                 float* __restrict__ xout)
{
  int t = blockIdx.x * 256 + threadIdx.x;
  int s = t / NSTREAM;             // 0 = posture, 1 = motion (wave-uniform)
  int m = t - s * NSTREAM;
  int n = 2 * m + s;

  const float* w1 = s ? mw1 : pw1;
  const float* b1 = s ? mb1 : pb1;
  const float* w2 = s ? mw2 : pw2;
  const float* b2 = s ? mb2 : pb2;
  const float  a  = s ? ma[0] : pa[0];

  const float* hin = h + (long)n * 192;
  float*       xo  = xout + (long)n * FDIM;

  for (int g = 0; g < 3; ++g) {
    float inb[64];
#pragma unroll
    for (int i = 0; i < 64; ++i) inb[i] = hin[g * 64 + i];
    float acc[32];
#pragma unroll
    for (int o = 0; o < 32; ++o) acc[o] = b2[g * 32 + o];

    for (int i = 0; i < 64; ++i) {
      const float* wr = w1 + (long)(g * 64 + i) * 64;
      float sv = b1[g * 64 + i];
#pragma unroll
      for (int k = 0; k < 64; ++k) sv += wr[k] * inb[k];
      float mv = prelu_f(sv, a);
#pragma unroll
      for (int o = 0; o < 32; ++o) acc[o] += w2[(long)(g * 32 + o) * 64 + i] * mv;
    }
#pragma unroll
    for (int o = 0; o < 32; ++o) xo[g * 32 + o] = acc[o];
  }
}

// ---------------------------------------------------------------------------
// Kernel 2: edge gate  e = tanh( prelu((x[dst]*x[src]) @ W1 + b1, ga) @ w2 + b2 )
// One wave per 16-edge tile, 8 tiles/wave. A tile (f16) + W1^T (f16, staged
// once per block) in LDS; A-fragments hoisted to VGPRs; 6 N-tiles x 3 K-steps
// of v_wmma_f32_16x16x32_f16 with only B-fragment DS loads in the inner loop.
// ---------------------------------------------------------------------------
#define GAP 104   // A-tile row pitch (f16)
#define GWP 104   // W1^T row pitch (f16)
#define GHP 100   // hidden row pitch (f32)
#define GATE_TPW 8

__global__ __launch_bounds__(128)
void k_gate(const float* __restrict__ x,
            const int* __restrict__ src, const int* __restrict__ dst,
            const float* __restrict__ W1, const float* __restrict__ b1,
            const float* __restrict__ ga, const float* __restrict__ w2,
            const float* __restrict__ gb2,
            float* __restrict__ e_out)
{
  __shared__ __align__(16) h16 W1s[96 * GWP];
  __shared__ float b1s[96];
  __shared__ __align__(16) float w2s[96];
  __shared__ __align__(16) h16 As[4][16 * GAP];
  __shared__ __align__(16) float Hs[4][16 * GHP];

  const int tid = threadIdx.x;
  for (int idx = tid; idx < 96 * 96; idx += 128) {
    int i = idx / 96, o = idx - i * 96;      // W1[i][o] -> W1^T[o][i]
    W1s[o * GWP + i] = (h16)W1[idx];
  }
  if (tid < 96) { b1s[tid] = b1[tid]; w2s[tid] = w2[tid]; }
  __syncthreads();

  const float gav = ga[0];
  const float b2v = gb2[0];

  const int wid  = tid >> 5;
  const int lane = tid & 31;
  const int hl   = lane >> 4;   // lane half (0/1)
  const int row  = lane & 15;
  h16*   Aw = As[wid];
  float* Hw = Hs[wid];

  const int tile0 = (blockIdx.x * 4 + wid) * GATE_TPW;
  for (int it = 0; it < GATE_TPW; ++it) {
    const int tile = tile0 + it;
    const long j = (long)tile * 16 + row;
    const int sj = src[j], dj = dst[j];
    const float* xs = x + (long)sj * FDIM + hl * 48;
    const float* xd = x + (long)dj * FDIM + hl * 48;
#pragma unroll
    for (int c = 0; c < 48; c += 4) {
      float4 aa = *(const float4*)(xs + c);
      float4 bb = *(const float4*)(xd + c);
      int cc = row * GAP + hl * 48 + c;
      Aw[cc + 0] = (h16)(aa.x * bb.x);
      Aw[cc + 1] = (h16)(aa.y * bb.y);
      Aw[cc + 2] = (h16)(aa.z * bb.z);
      Aw[cc + 3] = (h16)(aa.w * bb.w);
    }

    // hoist A K-fragments into VGPRs (one tile row, 3 K-steps)
    const h16* Arow = Aw + row * GAP;        // A: M = lane&15 for both halves
    v16h af[3];
#pragma unroll
    for (int kt = 0; kt < 3; ++kt) {
      int ka = kt * 32 + hl * 8;             // A: K 0-7/16-23 vs 8-15/24-31
      v8h a0 = *(const v8h*)(Arow + ka);
      v8h a1 = *(const v8h*)(Arow + ka + 16);
      af[kt] = mk16(a0, a1);
    }

    for (int nt = 0; nt < 6; ++nt) {
      const int col = nt * 16 + row;         // B/C: N = lane&15
      const h16* Brow = W1s + col * GWP;
      v8f acc = {};
#pragma unroll
      for (int kt = 0; kt < 3; ++kt) {
        int kb = kt * 32 + hl * 8;           // B: K 0-7/16-23 vs 8-15/24-31
        v8h b0 = *(const v8h*)(Brow + kb);
        v8h b1f = *(const v8h*)(Brow + kb + 16);
        acc = __builtin_amdgcn_wmma_f32_16x16x32_f16(
            false, af[kt], false, mk16(b0, b1f), (short)0, acc, false, false);
      }
      const float bias = b1s[col];
#pragma unroll
      for (int r = 0; r < 8; ++r)            // C: M = r + 8*half
        Hw[(r + hl * 8) * GHP + col] = prelu_f(acc[r] + bias, gav);
    }

    // per-edge 96-dot with w2; two lane-halves each sum 48, combine via shfl
    float p = 0.f;
#pragma unroll
    for (int c = 0; c < 48; c += 4) {
      float4 hv = *(const float4*)(&Hw[row * GHP + hl * 48 + c]);
      float4 wv = *(const float4*)(&w2s[hl * 48 + c]);
      p += hv.x * wv.x + hv.y * wv.y + hv.z * wv.z + hv.w * wv.w;
    }
    p += __shfl_xor(p, 16);
    if (hl == 0)
      e_out[(long)tile * 16 + row] = tanhf(p + b2v);
  }
}

// ---------------------------------------------------------------------------
// Kernel 3: z[dst] += x[src] * e   (z pre-initialized to x, so z = x + segsum)
// One wave per edge; lane covers channels {lane, lane+32, lane+64}.
// ---------------------------------------------------------------------------
__global__ __launch_bounds__(256)
void k_scatter(const float* __restrict__ x, const int* __restrict__ src,
               const int* __restrict__ dst, const float* __restrict__ e,
               float* __restrict__ z)
{
  const long edge = (long)((blockIdx.x * 256 + threadIdx.x) >> 5);
  const int lane = threadIdx.x & 31;
  const int sj = src[edge], dj = dst[edge];
  const float ev = e[edge];
  const float* xr = x + (long)sj * FDIM;
  float*       zr = z + (long)dj * FDIM;
  atomicAdd(&zr[lane],      xr[lane]      * ev);
  atomicAdd(&zr[lane + 32], xr[lane + 32] * ev);
  atomicAdd(&zr[lane + 64], xr[lane + 64] * ev);
}

// ---------------------------------------------------------------------------
// Kernel 4: node update  x' = prelu( BN( LN(z) ) ).  One wave per node.
// ---------------------------------------------------------------------------
__global__ __launch_bounds__(256)
void k_node(const float* __restrict__ z,
            const float* __restrict__ lng, const float* __restrict__ lnb,
            const float* __restrict__ bng, const float* __restrict__ bnb,
            const float* __restrict__ bnrm, const float* __restrict__ bnrv,
            const float* __restrict__ acta,
            float* __restrict__ xo)
{
  const int node = (blockIdx.x * 256 + threadIdx.x) >> 5;
  const int lane = threadIdx.x & 31;
  const float aa = acta[0];
  const float* zr = z + (long)node * FDIM;
  float v0 = zr[lane], v1 = zr[lane + 32], v2 = zr[lane + 64];

  float s = v0 + v1 + v2;
#pragma unroll
  for (int off = 16; off; off >>= 1) s += __shfl_xor(s, off);
  const float mean = s * (1.f / 96.f);
  const float d0 = v0 - mean, d1 = v1 - mean, d2 = v2 - mean;
  float q = d0 * d0 + d1 * d1 + d2 * d2;
#pragma unroll
  for (int off = 16; off; off >>= 1) q += __shfl_xor(q, off);
  const float inv = rsqrtf(q * (1.f / 96.f) + 1e-5f);

  float* xr = xo + (long)node * FDIM;
  const float dd[3] = {d0, d1, d2};
#pragma unroll
  for (int k = 0; k < 3; ++k) {
    const int c = lane + k * 32;
    float ln = dd[k] * inv * lng[c] + lnb[c];
    float bn = (ln - bnrm[c]) * rsqrtf(bnrv[c] + 1e-5f) * bng[c] + bnb[c];
    xr[c] = prelu_f(bn, aa);
  }
}

// ---------------------------------------------------------------------------
// Kernel 5: out[N,128] = [raw | x1 | x2] @ t2_w + t2_b  via WMMA.
// Block = 4 waves x 16-node tiles; A K-fragments (9 x v16h) preloaded into
// VGPRs and held across the N-tile loop; W-tile (16 cols x 288) transposed-
// staged per N-tile in LDS.
// ---------------------------------------------------------------------------
#define OAP 296
__global__ __launch_bounds__(128)
void k_out(const float* __restrict__ x0, const float* __restrict__ x1,
           const float* __restrict__ x2, const float* __restrict__ tw,
           const float* __restrict__ tb, float* __restrict__ out)
{
  __shared__ __align__(16) h16 As[4][16 * OAP];
  __shared__ __align__(16) h16 WTs[16 * OAP];

  const int tid = threadIdx.x, wid = tid >> 5, lane = tid & 31;
  const int hl = lane >> 4, row = lane & 15;
  const int nodebase = blockIdx.x * 64 + wid * 16;
  const int n = nodebase + row;

  { // stage A: concat [x0 | x1 | x2] row of node n, this lane covers 144 cols
    const float* r0 = x0 + (long)n * FDIM;
    const float* r1 = x1 + (long)n * FDIM;
    const float* r2 = x2 + (long)n * FDIM;
    h16* Ar = As[wid] + row * OAP;
#pragma unroll
    for (int c4 = 0; c4 < 144; c4 += 4) {
      int cc = hl * 144 + c4;
      const float* sp = cc < 96 ? (r0 + cc) : (cc < 192 ? (r1 + cc - 96) : (r2 + cc - 192));
      float4 v = *(const float4*)sp;
      Ar[cc + 0] = (h16)v.x; Ar[cc + 1] = (h16)v.y;
      Ar[cc + 2] = (h16)v.z; Ar[cc + 3] = (h16)v.w;
    }
  }

  // hoist all 9 A K-fragments into VGPRs (held across the nt loop)
  const h16* Arow = As[wid] + row * OAP;
  v16h af[9];
#pragma unroll
  for (int kt = 0; kt < 9; ++kt) {
    int ka = kt * 32 + hl * 8;
    v8h a0 = *(const v8h*)(Arow + ka);
    v8h a1 = *(const v8h*)(Arow + ka + 16);
    af[kt] = mk16(a0, a1);
  }

  for (int nt = 0; nt < 8; ++nt) {
    __syncthreads();
    for (int idx = tid; idx < 16 * 288; idx += 128) {
      int nn = idx / 288, k = idx - nn * 288;     // WT[nn][k] = tw[k][nt*16+nn]
      WTs[nn * OAP + k] = (h16)tw[(long)k * OUTF + nt * 16 + nn];
    }
    __syncthreads();

    const h16* Brow = WTs + row * OAP;
    v8f acc = {};
#pragma unroll
    for (int kt = 0; kt < 9; ++kt) {
      int kb = kt * 32 + hl * 8;
      v8h b0 = *(const v8h*)(Brow + kb);
      v8h b1f = *(const v8h*)(Brow + kb + 16);
      acc = __builtin_amdgcn_wmma_f32_16x16x32_f16(
          false, af[kt], false, mk16(b0, b1f), (short)0, acc, false, false);
    }
    const int col = nt * 16 + row;
    const float bias = tb[col];
#pragma unroll
    for (int r = 0; r < 8; ++r)
      out[(long)(nodebase + r + hl * 8) * OUTF + col] = acc[r] + bias;
  }
}

// ---------------------------------------------------------------------------
// Launcher
// ---------------------------------------------------------------------------
extern "C" void kernel_launch(void* const* d_in, const int* in_sizes, int n_in,
                              void* d_out, int out_size, void* d_ws, size_t ws_size,
                              hipStream_t stream)
{
  const float* h     = (const float*)d_in[0];
  const int*   src   = (const int*)  d_in[1];
  const int*   dst   = (const int*)  d_in[2];
  const float* tp_w1 = (const float*)d_in[3];
  const float* tp_b1 = (const float*)d_in[4];
  const float* tp_a  = (const float*)d_in[5];
  const float* tp_w2 = (const float*)d_in[6];
  const float* tp_b2 = (const float*)d_in[7];
  const float* tm_w1 = (const float*)d_in[8];
  const float* tm_b1 = (const float*)d_in[9];
  const float* tm_a  = (const float*)d_in[10];
  const float* tm_w2 = (const float*)d_in[11];
  const float* tm_b2 = (const float*)d_in[12];
  const float* ln_g  = (const float*)d_in[13];
  const float* ln_b  = (const float*)d_in[14];
  const float* bn_g  = (const float*)d_in[15];
  const float* bn_b  = (const float*)d_in[16];
  const float* bn_rm = (const float*)d_in[17];
  const float* bn_rv = (const float*)d_in[18];
  const float* act_a = (const float*)d_in[19];
  const float* gW1   = (const float*)d_in[20];
  const float* gb1   = (const float*)d_in[21];
  const float* ga    = (const float*)d_in[22];
  const float* gW2   = (const float*)d_in[23];
  const float* gb2   = (const float*)d_in[24];
  const float* t2_w  = (const float*)d_in[25];
  const float* t2_b  = (const float*)d_in[26];

  float* out = (float*)d_out;                       // [25600*128]
  float* ee  = out + (long)N_NODES * OUTF;          // [2*E]

  float* x0 = (float*)d_ws;                         // raw features  [N,96]
  float* x1 = x0 + NFLT;
  float* x2 = x1 + NFLT;
  float* z  = x2 + NFLT;
  float* eb = z  + NFLT;                            // aggregation gate [E]

  // 1) input transform -> x0 (raw)
  k_transform<<<100, 256, 0, stream>>>(h, tp_w1, tp_b1, tp_a, tp_w2, tp_b2,
                                       tm_w1, tm_b1, tm_a, tm_w2, tm_b2, x0);

  // --- layer 0 ---
  hipMemcpyAsync(z, x0, (size_t)NFLT * sizeof(float), hipMemcpyDeviceToDevice, stream);
  k_gate<<<800, 128, 0, stream>>>(x0, src, dst, gW1, gb1, ga, gW2, gb2, eb);
  k_scatter<<<NE / 8, 256, 0, stream>>>(x0, src, dst, eb, z);
  k_node<<<N_NODES / 8, 256, 0, stream>>>(z, ln_g, ln_b, bn_g, bn_b, bn_rm, bn_rv,
                                          act_a, x1);
  k_gate<<<800, 128, 0, stream>>>(x1, src, dst, gW1, gb1, ga, gW2, gb2, ee);

  // --- layer 1 (parameter offsets: W1 +96*96, vectors +96, scalars +1) ---
  const float* gW1b = gW1 + 96 * 96;
  hipMemcpyAsync(z, x1, (size_t)NFLT * sizeof(float), hipMemcpyDeviceToDevice, stream);
  k_gate<<<800, 128, 0, stream>>>(x1, src, dst, gW1b, gb1 + 96, ga + 1, gW2 + 96,
                                  gb2 + 1, eb);
  k_scatter<<<NE / 8, 256, 0, stream>>>(x1, src, dst, eb, z);
  k_node<<<N_NODES / 8, 256, 0, stream>>>(z, ln_g + 96, ln_b + 96, bn_g + 96,
                                          bn_b + 96, bn_rm + 96, bn_rv + 96,
                                          act_a + 1, x2);
  k_gate<<<800, 128, 0, stream>>>(x2, src, dst, gW1b, gb1 + 96, ga + 1, gW2 + 96,
                                  gb2 + 1, ee + NE);

  // 5) final projection
  k_out<<<400, 128, 0, stream>>>(x0, x1, x2, t2_w, t2_b, out);
}
